// CNN_22952305230196
// MI455X (gfx1250) — matrix operand order repaired and checked by reading, not compiled, not run on hardware
//
#include <hip/hip_runtime.h>
#include <hip/hip_bf16.h>

typedef __attribute__((ext_vector_type(16))) _Float16 v16h;
typedef __attribute__((ext_vector_type(8)))  _Float16 v8h;
typedef __attribute__((ext_vector_type(8)))  float    v8f;
typedef __attribute__((ext_vector_type(4)))  unsigned int u32x4;
typedef __attribute__((ext_vector_type(8)))  int      i32x8;
typedef __attribute__((ext_vector_type(4)))  int      i32x4;

#if defined(__has_builtin)
#if __has_builtin(__builtin_amdgcn_tensor_load_to_lds) && __has_builtin(__builtin_amdgcn_s_wait_tensorcnt)
#define HAVE_TDM 1
#else
#define HAVE_TDM 0
#endif
#else
#define HAVE_TDM 0
#endif

// h1 is stored zero-PADDED: (32, 4, 258, 258) f16, interior at [y+1][x+1].
#define H1_W      258
#define H1_CHSTR  (258 * 258)        // 66564
#define H1_IMSTR  (4 * H1_CHSTR)     // 266256
#define H1_ELEMS  (32 * H1_IMSTR)    // 8,520,192 halves

// ---------------------------------------------------------------------------
// Kernel 0: zero-fill padded h1 (so conv2 never needs bounds checks).
// 4,260,096 dwords = 16641 blocks * 256 exactly.
// ---------------------------------------------------------------------------
__global__ void nv_zero_h1(unsigned int* __restrict__ h1w) {
  h1w[blockIdx.x * blockDim.x + threadIdx.x] = 0u;
}

// ---------------------------------------------------------------------------
// Kernel 1: conv1 (3->4, 3x3, SAME) + bias + ReLU + 2x2 avgpool -> padded f16
// One thread per pooled interior element (32,4,256,256) = 2^23 threads.
// ---------------------------------------------------------------------------
__global__ void nv_conv1_relu_pool(const float* __restrict__ x,      // (32,3,512,512)
                                   const float* __restrict__ w,      // (4,3,3,3)
                                   const float* __restrict__ b,      // (4,)
                                   _Float16* __restrict__ h1p) {     // padded
  int idx = blockIdx.x * blockDim.x + threadIdx.x;
  int xo = idx & 255;
  int yo = (idx >> 8) & 255;
  int co = (idx >> 16) & 3;
  int n  = idx >> 18;
  const float* xn = x + (size_t)n * 3 * 512 * 512;
  float bias = b[co];
  float acc = 0.f;
#pragma unroll
  for (int iy = 0; iy < 2; ++iy) {
#pragma unroll
    for (int ix = 0; ix < 2; ++ix) {
      int py = 2 * yo + iy, px = 2 * xo + ix;
      float s = bias;
#pragma unroll
      for (int ci = 0; ci < 3; ++ci) {
        const float* xc = xn + (size_t)ci * 512 * 512;
        const float* wc = w + (co * 3 + ci) * 9;
#pragma unroll
        for (int ky = 0; ky < 3; ++ky) {
          int yy = py + ky - 1;
          if (yy < 0 || yy >= 512) continue;
#pragma unroll
          for (int kx = 0; kx < 3; ++kx) {
            int xx = px + kx - 1;
            if (xx < 0 || xx >= 512) continue;
            s += wc[ky * 3 + kx] * xc[yy * 512 + xx];
          }
        }
      }
      acc += fmaxf(s, 0.f);
    }
  }
  h1p[(size_t)(n * 4 + co) * H1_CHSTR + (yo + 1) * H1_W + (xo + 1)] =
      (_Float16)(0.25f * acc);
}

// ---------------------------------------------------------------------------
// Kernel 2: conv2 (4->16) implicit GEMM on v_wmma_f32_16x16x32_f16 + ReLU +
// 2x2 avgpool. Block = 8 waves = a 2-row x 128-col strip of one image.
// Input tile (4ch x 4rows x 130cols f16) staged in LDS via TDM
// (tensor_load_to_lds + s_wait_tensorcnt) with a cooperative-copy fallback.
// K-map: Kg = ci*16 + tap (tap = (dy+1)*3 + (dx+1), taps 9..15 are zero pad).
// All operand reads are branch-free (padded h1, padded weights).
// ---------------------------------------------------------------------------
__global__ void nv_conv2_wmma_pool(const _Float16* __restrict__ h1p,
                                   const float* __restrict__ w,      // (16,4,3,3)
                                   const float* __restrict__ bias,   // (16,)
                                   float* __restrict__ h2) {         // (32,16,128,128)
  __shared__ _Float16 tile[4 * 4 * 130];   // [ci][yy 0..3][xx 0..129]
  __shared__ _Float16 wlds[16 * 4 * 16];   // [co][ci][tap padded to 16]
  int tid = threadIdx.x;
  int xhalf = blockIdx.x & 1;
  int rp = (blockIdx.x >> 1) & 127;
  int n  = blockIdx.x >> 8;
  int X0 = xhalf * 128;   // padded x of tile col 0 (== interior x0-1)
  int Y0 = rp * 2;        // padded y of tile row 0

  // stage weights as fragment-ready f16 (tap padded to 16)
  for (int i = tid; i < 1024; i += 256) {
    int tap = i & 15;
    wlds[i] = (tap < 9) ? (_Float16)w[(i >> 4) * 9 + tap] : (_Float16)0.f;
  }

  const _Float16* hn = h1p + (size_t)n * H1_IMSTR;
#if HAVE_TDM
  if (tid < 32) {
    unsigned long long ga =
        (unsigned long long)(size_t)hn + (unsigned long long)((Y0 * H1_W + X0) * 2);
    unsigned int lds_off = (unsigned int)(size_t)(&tile[0]);
    // D# group 0: count=1, lds_addr, global_addr(57b), type=2
    u32x4 g0 = {1u, lds_off, (unsigned int)ga,
                (((unsigned int)(ga >> 32)) & 0x01FFFFFFu) | 0x80000000u};
    // D# group 1: data_size=2B; tensor_dim0/1 = 258; tile = 130 x 4 x 4;
    //             dim0_stride = 258; dim1_stride = 66564 (0x10404)
    i32x8 g1 = {(int)(1u << 16),
                (int)(258u << 16),
                (int)(258u << 16),
                (int)(130u << 16),
                (int)((4u << 16) | 4u),
                (int)258,
                (int)(0x0404u << 16),
                (int)1};
    // D# group 2: tensor_dim2 = 4 (channels); dims 3+ unused
    i32x4 g2 = {4, 0, 0, 0};
    i32x4 g3 = {0, 0, 0, 0};
    // 6-arg toolchain form: extra 256-bit group, zero-filled; cpol = 0
    i32x8 g4 = {0, 0, 0, 0, 0, 0, 0, 0};
    __builtin_amdgcn_tensor_load_to_lds(g0, g1, g2, g3, g4, 0);
    __builtin_amdgcn_s_wait_tensorcnt(0);
  }
#else
  for (int i = tid; i < 4 * 4 * 130; i += 256) {
    int ci = i / 520, rem = i % 520;
    int yy = rem / 130, xx = rem % 130;
    tile[i] = hn[(size_t)ci * H1_CHSTR + (Y0 + yy) * H1_W + (X0 + xx)];
  }
#endif
  __syncthreads();

  int lane = tid & 31;
  int wv = tid >> 5;
  int hh = lane >> 4;       // lane half
  int Mrow = lane & 15;     // A: output channel; B: pixel column in 16-tile
  int bx = wv * 16 + Mrow;  // local pixel x within the 128-col strip

  // ---- A fragments: wlds half-index = Mrow*64 + Kg; A 16x32 k order used here
  //   half i -> Kg = 16*hh + i (contiguous order, matches B)
  // => two contiguous aligned 8-half runs per fragment.
  int abase = Mrow * 64 + 16 * hh;
  v8h al0 = *(const v8h*)&wlds[abase];
  v8h al1 = *(const v8h*)&wlds[abase + 8];
  v8h ah0 = *(const v8h*)&wlds[abase + 32];
  v8h ah1 = *(const v8h*)&wlds[abase + 40];
  v16h a_lo, a_hi;
#pragma unroll
  for (int i = 0; i < 8; ++i) {
    a_lo[i] = al0[i]; a_lo[8 + i] = al1[i];
    a_hi[i] = ah0[i]; a_hi[8 + i] = ah1[i];
  }

  // ---- per-row B fragments (branch-free LDS reads) + WMMA
  v8f cr[2];
#pragma unroll
  for (int row = 0; row < 2; ++row) {
    v16h b_lo = {}, b_hi = {};
#pragma unroll
    for (int i = 0; i < 9; ++i) {       // tap i -> dy=i/3, dx=i%3 (already +1)
      int yy = row + i / 3;
      int xx = bx + i % 3;
      b_lo[i] = tile[(hh * 4 + yy) * 130 + xx];        // ci = hh
      b_hi[i] = tile[((2 + hh) * 4 + yy) * 130 + xx];  // ci = 2+hh
    }
    v8f c = {};
    c = __builtin_amdgcn_wmma_f32_16x16x32_f16(false, a_lo, false, b_lo,
                                               (short)0, c, false, false);
    c = __builtin_amdgcn_wmma_f32_16x16x32_f16(false, a_hi, false, b_hi,
                                               (short)0, c, false, false);
    cr[row] = c;
  }

  // ---- bias + ReLU + 2x2 avgpool (x-pairs via shfl_xor(1))
  float* outp = h2 + (size_t)n * 16 * 16384;
#pragma unroll
  for (int r = 0; r < 8; ++r) {
    int co = r + 8 * hh;                 // C/D layout: VGPR r = rows r / r+8
    float bco = bias[co];
    float v0 = fmaxf(cr[0][r] + bco, 0.f);
    float v1 = fmaxf(cr[1][r] + bco, 0.f);
    float v0x = __shfl_xor(v0, 1, 32);
    float v1x = __shfl_xor(v1, 1, 32);
    float pooled = 0.25f * (v0 + v0x + v1 + v1x);
    if ((lane & 1) == 0) {
      int ox = 64 * xhalf + 8 * wv + (Mrow >> 1);
      outp[((size_t)co * 128 + rp) * 128 + ox] = pooled;
    }
  }
}

// ---------------------------------------------------------------------------
// Kernel 3: NetVLAD soft-assign + aggregation. One block (256 threads)/image.
// ---------------------------------------------------------------------------
__global__ void nv_vlad_accum(const float* __restrict__ h2,    // (32,16,128,128)
                              const float* __restrict__ aw,    // (4,16)
                              const float* __restrict__ ab,    // (4,)
                              float* __restrict__ vlad,        // (32,64)
                              float* __restrict__ asum) {      // (32,4)
  int n = blockIdx.x;
  int tid = threadIdx.x;
  const float* xn = h2 + (size_t)n * 16 * 16384;

  __shared__ float sW[64];
  __shared__ float sB[4];
  if (tid < 64) sW[tid] = aw[tid];
  if (tid < 4)  sB[tid] = ab[tid];
  __syncthreads();

  float acc[64];
#pragma unroll
  for (int i = 0; i < 64; ++i) acc[i] = 0.f;
  float as[4] = {0.f, 0.f, 0.f, 0.f};

  for (int m = tid; m < 16384; m += 256) {
    float xv[16];
#pragma unroll
    for (int c = 0; c < 16; ++c) xv[c] = xn[(size_t)c * 16384 + m];
    float l[4];
#pragma unroll
    for (int k = 0; k < 4; ++k) {
      float s = sB[k];
#pragma unroll
      for (int c = 0; c < 16; ++c) s += sW[k * 16 + c] * xv[c];
      l[k] = s;
    }
    float mx = fmaxf(fmaxf(l[0], l[1]), fmaxf(l[2], l[3]));
    float e[4], ssum = 0.f;
#pragma unroll
    for (int k = 0; k < 4; ++k) { e[k] = __expf(l[k] - mx); ssum += e[k]; }
    float inv = 1.f / ssum;
#pragma unroll
    for (int k = 0; k < 4; ++k) {
      float a = e[k] * inv;
      as[k] += a;
#pragma unroll
      for (int c = 0; c < 16; ++c) acc[k * 16 + c] += a * xv[c];
    }
  }

#pragma unroll
  for (int off = 16; off > 0; off >>= 1) {
#pragma unroll
    for (int i = 0; i < 64; ++i) acc[i] += __shfl_down(acc[i], off, 32);
#pragma unroll
    for (int k = 0; k < 4; ++k) as[k] += __shfl_down(as[k], off, 32);
  }

  __shared__ float part[8][68];
  int wvi = tid >> 5, ln = tid & 31;
  if (ln == 0) {
#pragma unroll
    for (int i = 0; i < 64; ++i) part[wvi][i] = acc[i];
#pragma unroll
    for (int k = 0; k < 4; ++k) part[wvi][64 + k] = as[k];
  }
  __syncthreads();
  if (tid < 68) {
    float s = 0.f;
#pragma unroll
    for (int w2 = 0; w2 < 8; ++w2) s += part[w2][tid];
    if (tid < 64) vlad[n * 64 + tid] = s;
    else          asum[n * 4 + (tid - 64)] = s;
  }
}

// ---------------------------------------------------------------------------
// Kernel 4: residual, intra L2 norm, global L2 norm, 64x7 linear head.
// ---------------------------------------------------------------------------
__global__ void nv_finalize(const float* __restrict__ vlad,   // (32,64)
                            const float* __restrict__ asum,   // (32,4)
                            const float* __restrict__ cent,   // (4,16)
                            const float* __restrict__ lw,     // (7,64)
                            const float* __restrict__ lb,     // (7,)
                            float* __restrict__ out) {        // (32,7)
  int n = blockIdx.x;
  int t = threadIdx.x;            // 64 threads
  int k = t >> 4, c = t & 15;
  float v = vlad[n * 64 + t] - asum[n * 4 + k] * cent[k * 16 + c];

  float s = v * v;
#pragma unroll
  for (int off = 1; off < 16; off <<= 1) s += __shfl_xor(s, off, 16);
  v = v / fmaxf(sqrtf(s), 1e-12f);

  __shared__ float sh[64];
  __shared__ float stot;
  sh[t] = v * v;
  __syncthreads();
  if (t == 0) {
    float s2 = 0.f;
#pragma unroll
    for (int i = 0; i < 64; ++i) s2 += sh[i];
    stot = fmaxf(sqrtf(s2), 1e-12f);
  }
  __syncthreads();
  v /= stot;
  sh[t] = v;
  __syncthreads();
  if (t < 7) {
    float o = lb[t];
#pragma unroll
    for (int i = 0; i < 64; ++i) o += lw[t * 64 + i] * sh[i];
    out[n * 7 + t] = o;
  }
}

// ---------------------------------------------------------------------------
extern "C" void kernel_launch(void* const* d_in, const int* in_sizes, int n_in,
                              void* d_out, int out_size, void* d_ws, size_t ws_size,
                              hipStream_t stream) {
  (void)in_sizes; (void)n_in; (void)out_size; (void)ws_size;
  const float* x    = (const float*)d_in[0];
  const float* c1w  = (const float*)d_in[1];
  const float* c1b  = (const float*)d_in[2];
  const float* c2w  = (const float*)d_in[3];
  const float* c2b  = (const float*)d_in[4];
  const float* cent = (const float*)d_in[5];
  const float* aw   = (const float*)d_in[6];
  const float* ab   = (const float*)d_in[7];
  const float* lw   = (const float*)d_in[8];
  const float* lb   = (const float*)d_in[9];
  float* out = (float*)d_out;

  char* ws = (char*)d_ws;
  const size_t H1_BYTES = (size_t)H1_ELEMS * sizeof(_Float16);           // 17.0 MB
  const size_t H2_BYTES = (size_t)32 * 16 * 128 * 128 * sizeof(float);   // 33.6 MB
  _Float16* h1p  = (_Float16*)ws;
  float*    h2   = (float*)(ws + H1_BYTES);
  float*    vlad = (float*)(ws + H1_BYTES + H2_BYTES);
  float*    asum = (float*)(ws + H1_BYTES + H2_BYTES + (size_t)32 * 64 * sizeof(float));

  // zero the padded h1 (border stays zero): 4,260,096 dwords = 16641*256
  nv_zero_h1<<<16641, 256, 0, stream>>>((unsigned int*)h1p);
  // conv1: 2^23 interior elements
  nv_conv1_relu_pool<<<32768, 256, 0, stream>>>(x, c1w, c1b, h1p);
  // conv2: 32 n * 128 row-pairs * 2 col-halves = 8192 blocks
  nv_conv2_wmma_pool<<<8192, 256, 0, stream>>>(h1p, c2w, c2b, h2);
  // NetVLAD: one block per image
  nv_vlad_accum<<<32, 256, 0, stream>>>(h2, aw, ab, vlad, asum);
  // head
  nv_finalize<<<32, 64, 0, stream>>>(vlad, asum, cent, lw, lb, out);
}
